// StructuralEncoder_81913616269478
// MI455X (gfx1250) — matrix operand order repaired and checked by reading, not compiled, not run on hardware
//
#include <hip/hip_runtime.h>
#include <hip/hip_bf16.h>

#define NNODES 50000
#define NEDGES 1600000
#define HD 128
#define CD 64
#define NL 3

typedef __attribute__((ext_vector_type(16))) __bf16 v16bf;
typedef __attribute__((ext_vector_type(8)))  float  v8f;

union FragBF { uint4 u[2]; v16bf v; };

__device__ __forceinline__ unsigned short f2bf(float f) {
    unsigned int u = __float_as_uint(f);
    unsigned int r = u + 0x7FFFu + ((u >> 16) & 1u);   // round-to-nearest-even
    return (unsigned short)(r >> 16);
}

// ---------------------------------------------------------------------------
// Weight pre-transpose: f32 [K][Nc] -> bf16 [Nc][K]  (B-operand wants K-contig)
// ---------------------------------------------------------------------------
__global__ void k_transpose_bf(const float* __restrict__ src,
                               unsigned short* __restrict__ dst,
                               int K, int Nc) {
    int i = blockIdx.x * blockDim.x + threadIdx.x;
    if (i >= K * Nc) return;
    int k = i / Nc, n = i - k * Nc;
    dst[(size_t)n * K + k] = f2bf(src[i]);
}

// ---------------------------------------------------------------------------
// Edge encoder: ef = ReLU(ea @ W1 + b1) @ W2 + b2   -> bf16 [E,64]
// ---------------------------------------------------------------------------
__global__ __launch_bounds__(256)
void k_edge_enc(const float* __restrict__ ea,
                const float* __restrict__ w1, const float* __restrict__ b1,
                const float* __restrict__ w2, const float* __restrict__ b2,
                unsigned short* __restrict__ ef) {
    __shared__ float sW1[3 * 64], sB1[64], sW2[64 * 64], sB2[64];
    const int tid = threadIdx.x;
    for (int i = tid; i < 3 * 64; i += 256) sW1[i] = w1[i];
    for (int i = tid; i < 64; i += 256) { sB1[i] = b1[i]; sB2[i] = b2[i]; }
    for (int i = tid; i < 64 * 64; i += 256) sW2[i] = w2[i];
    __syncthreads();
    const int e = blockIdx.x * 256 + tid;
    if (e >= NEDGES) return;
    const float a0 = ea[(size_t)e * 3 + 0];
    const float a1 = ea[(size_t)e * 3 + 1];
    const float a2 = ea[(size_t)e * 3 + 2];
    float t[64];
#pragma unroll
    for (int c = 0; c < 64; ++c) {
        float v = sB1[c] + a0 * sW1[c] + a1 * sW1[64 + c] + a2 * sW1[128 + c];
        t[c] = v > 0.f ? v : 0.f;
    }
    for (int c2 = 0; c2 < 64; ++c2) {
        float acc = sB2[c2];
#pragma unroll
        for (int c = 0; c < 64; ++c) acc += t[c] * sW2[c * 64 + c2];
        ef[(size_t)e * 64 + c2] = f2bf(acc);
    }
}

// ---------------------------------------------------------------------------
// Geometry encoder: nf[:, :64] = ReLU(geo @ W + b), nf[:, 64:] = 0
// Writes f32 master nf and bf16 copy into xu[:, 0:128]
// ---------------------------------------------------------------------------
__global__ void k_geom(const float* __restrict__ geo,
                       const float* __restrict__ gw, const float* __restrict__ gb,
                       float* __restrict__ nf, unsigned short* __restrict__ xu) {
    const int node = blockIdx.x * blockDim.x + threadIdx.x;
    if (node >= NNODES) return;
    float g[5];
#pragma unroll
    for (int k = 0; k < 5; ++k) g[k] = geo[(size_t)node * 5 + k];
    for (int c = 0; c < 64; ++c) {
        float v = gb[c];
#pragma unroll
        for (int k = 0; k < 5; ++k) v += g[k] * gw[k * 64 + c];
        v = v > 0.f ? v : 0.f;
        nf[(size_t)node * 128 + c]      = v;
        nf[(size_t)node * 128 + 64 + c] = 0.f;
        xu[(size_t)node * 256 + c]      = f2bf(v);
        xu[(size_t)node * 256 + 64 + c] = 0;   // bf16 zero
    }
}

__global__ void k_zero(float* __restrict__ p, int n) {
    int i = blockIdx.x * blockDim.x + threadIdx.x;
    if (i < n) p[i] = 0.f;
}

// agg (f32) -> xu[:, 128:256] (bf16) for the update GEMM
__global__ void k_cvt_agg(const float* __restrict__ agg, unsigned short* __restrict__ xu) {
    int i = blockIdx.x * blockDim.x + threadIdx.x;
    if (i >= NNODES * 128) return;
    int node = i >> 7, f = i & 127;
    xu[(size_t)node * 256 + 128 + f] = f2bf(agg[i]);
}

// ---------------------------------------------------------------------------
// Message kernel: per wave 2 M-tiles (32 edges) x 128 outputs, fully WMMA.
// The j (output-tile) loop is split into 2 halves of 4 tiles so only 64
// accumulator VGPRs are live at a time -> compiler can batch-load 4 B frags
// (one clause, one wait) and issue 8 WMMAs back-to-back per k-step.
//   GEMM1: K=320 from gathered [nf[src]|nf[dst]|ef] bf16; ReLU+bias -> LDS bf16
//   GEMM2: K=128 from LDS; bias; atomicAdd scatter into agg[dst]
// ---------------------------------------------------------------------------
#define MSG_WAVES 4
__global__ __launch_bounds__(128)
void k_message(const long long* __restrict__ eidx,          // [2][E] int64
               const unsigned short* __restrict__ xu,       // [N][256] bf16 (first 128 = nf)
               const unsigned short* __restrict__ ef,       // [E][64]  bf16
               const unsigned short* __restrict__ w1t,      // [128][320] bf16 (W1^T)
               const float* __restrict__ b1,
               const unsigned short* __restrict__ w2t,      // [128][128] bf16 (W2^T)
               const float* __restrict__ b2,
               float* __restrict__ agg) {
    __shared__ unsigned short sH[MSG_WAVES][32 * 128];      // 32 KB
    const int tid  = threadIdx.x;
    const int wave = tid >> 5;
    const int lane = tid & 31;
    const int m    = lane & 15;
    const int hi   = lane >> 4;          // lane half
    const int aOff = hi * 8;             // A-operand K offset for this half
    const int eBase = blockIdx.x * (MSG_WAVES * 32) + wave * 32;
    int e0 = eBase + m;       if (e0 >= NEDGES) e0 = NEDGES - 1;
    int e1 = eBase + 16 + m;  if (e1 >= NEDGES) e1 = NEDGES - 1;
    const long long s0 = eidx[e0];
    const long long d0 = eidx[(size_t)NEDGES + e0];
    const long long s1 = eidx[e1];
    const long long d1 = eidx[(size_t)NEDGES + e1];

    const v8f zf = {};

    // ---------------- GEMM1 (K = 320), two j-halves ----------------
#pragma unroll
    for (int half = 0; half < 2; ++half) {
        v8f acc0[4], acc1[4];
#pragma unroll
        for (int jj = 0; jj < 4; ++jj) { acc0[jj] = zf; acc1[jj] = zf; }

#pragma unroll
        for (int ks = 0; ks < 10; ++ks) {
            const int kb = ks * 32;
            const unsigned short *rowp0, *rowp1;
            if (kb < 128) {
                rowp0 = xu + (size_t)s0 * 256 + (kb + aOff);
                rowp1 = xu + (size_t)s1 * 256 + (kb + aOff);
            } else if (kb < 256) {
                rowp0 = xu + (size_t)d0 * 256 + (kb - 128 + aOff);
                rowp1 = xu + (size_t)d1 * 256 + (kb - 128 + aOff);
            } else {
                rowp0 = ef + (size_t)e0 * 64 + (kb - 256 + aOff);
                rowp1 = ef + (size_t)e1 * 64 + (kb - 256 + aOff);
            }
            FragBF a0, a1;
            a0.u[0] = *(const uint4*)(rowp0);
            a0.u[1] = *(const uint4*)(rowp0 + 16);
            a1.u[0] = *(const uint4*)(rowp1);
            a1.u[1] = *(const uint4*)(rowp1 + 16);

            FragBF bArr[4];
#pragma unroll
            for (int jj = 0; jj < 4; ++jj) {
                const int j = half * 4 + jj;
                const unsigned short* bp =
                    w1t + (size_t)(j * 16 + m) * 320 + kb + hi * 8;
                bArr[jj].u[0] = *(const uint4*)(bp);
                bArr[jj].u[1] = *(const uint4*)(bp + 16);
            }
#pragma unroll
            for (int jj = 0; jj < 4; ++jj) {
                acc0[jj] = __builtin_amdgcn_wmma_f32_16x16x32_bf16(
                    false, a0.v, false, bArr[jj].v, (short)0, acc0[jj], false, false);
                acc1[jj] = __builtin_amdgcn_wmma_f32_16x16x32_bf16(
                    false, a1.v, false, bArr[jj].v, (short)0, acc1[jj], false, false);
            }
        }

        // hidden = ReLU(acc + b1) -> LDS bf16, row-major [32][128] per wave
#pragma unroll
        for (int jj = 0; jj < 4; ++jj) {
            const int n = (half * 4 + jj) * 16 + m;
            const float bias = b1[n];
#pragma unroll
            for (int r = 0; r < 8; ++r) {
                const int mm = r + hi * 8;
                float v0 = acc0[jj][r] + bias;
                float v1 = acc1[jj][r] + bias;
                sH[wave][mm * 128 + n]        = f2bf(v0 > 0.f ? v0 : 0.f);
                sH[wave][(16 + mm) * 128 + n] = f2bf(v1 > 0.f ? v1 : 0.f);
            }
        }
    }
    __syncthreads();

    // ---------------- GEMM2 (K = 128), two j-halves + scatter ----------------
#pragma unroll
    for (int half = 0; half < 2; ++half) {
        v8f acd0[4], acd1[4];
#pragma unroll
        for (int jj = 0; jj < 4; ++jj) { acd0[jj] = zf; acd1[jj] = zf; }

#pragma unroll
        for (int ks = 0; ks < 4; ++ks) {
            const int kb = ks * 32;
            const unsigned short* hp0 = &sH[wave][m * 128 + kb + aOff];
            const unsigned short* hp1 = &sH[wave][(16 + m) * 128 + kb + aOff];
            FragBF a0, a1;
            a0.u[0] = *(const uint4*)(hp0);
            a0.u[1] = *(const uint4*)(hp0 + 16);
            a1.u[0] = *(const uint4*)(hp1);
            a1.u[1] = *(const uint4*)(hp1 + 16);

            FragBF bArr[4];
#pragma unroll
            for (int jj = 0; jj < 4; ++jj) {
                const int j = half * 4 + jj;
                const unsigned short* bp =
                    w2t + (size_t)(j * 16 + m) * 128 + kb + hi * 8;
                bArr[jj].u[0] = *(const uint4*)(bp);
                bArr[jj].u[1] = *(const uint4*)(bp + 16);
            }
#pragma unroll
            for (int jj = 0; jj < 4; ++jj) {
                acd0[jj] = __builtin_amdgcn_wmma_f32_16x16x32_bf16(
                    false, a0.v, false, bArr[jj].v, (short)0, acd0[jj], false, false);
                acd1[jj] = __builtin_amdgcn_wmma_f32_16x16x32_bf16(
                    false, a1.v, false, bArr[jj].v, (short)0, acd1[jj], false, false);
            }
        }

        // scatter-add this half's columns into agg[dst] (L2-resident)
#pragma unroll
        for (int r = 0; r < 8; ++r) {
            const int mm = r + hi * 8;
            const int ee0 = eBase + mm;
            if (ee0 < NEDGES) {
                const long long dd = eidx[(size_t)NEDGES + ee0];
#pragma unroll
                for (int jj = 0; jj < 4; ++jj) {
                    const int n = (half * 4 + jj) * 16 + m;
                    atomicAdd(agg + (size_t)dd * 128 + n, acd0[jj][r] + b2[n]);
                }
            }
            const int ee1 = eBase + 16 + mm;
            if (ee1 < NEDGES) {
                const long long dd = eidx[(size_t)NEDGES + ee1];
#pragma unroll
                for (int jj = 0; jj < 4; ++jj) {
                    const int n = (half * 4 + jj) * 16 + m;
                    atomicAdd(agg + (size_t)dd * 128 + n, acd1[jj][r] + b2[n]);
                }
            }
        }
    }
}

// ---------------------------------------------------------------------------
// Update kernel: new = ReLU([nf|agg] @ W1 + b1) @ W2 + b2 ; x = nf + new ;
// LayerNorm(x) -> nf (f32) and xu[:, :128] (bf16). 4 waves, 64 nodes/block.
// ---------------------------------------------------------------------------
__global__ __launch_bounds__(128)
void k_update(unsigned short* __restrict__ xu, float* __restrict__ nf,
              const unsigned short* __restrict__ w1t,   // [128][256]
              const float* __restrict__ b1,
              const unsigned short* __restrict__ w2t,   // [128][128]
              const float* __restrict__ b2,
              const float* __restrict__ lng, const float* __restrict__ lnb) {
    __shared__ unsigned short sH[4][16 * 128];
    __shared__ float          sX[4][16 * 128];
    const int tid  = threadIdx.x;
    const int wave = tid >> 5;
    const int lane = tid & 31;
    const int m    = lane & 15;
    const int hi   = lane >> 4;
    const int aOff = hi * 8;
    const int nBase = blockIdx.x * 64 + wave * 16;
    int node = nBase + m;
    if (node >= NNODES) node = NNODES - 1;

    const v8f zf = {};

#pragma unroll
    for (int half = 0; half < 2; ++half) {
        v8f acc[4];
#pragma unroll
        for (int jj = 0; jj < 4; ++jj) acc[jj] = zf;
#pragma unroll
        for (int ks = 0; ks < 8; ++ks) {
            const int kb = ks * 32;
            const unsigned short* rowp = xu + (size_t)node * 256 + kb + aOff;
            FragBF a;
            a.u[0] = *(const uint4*)(rowp);
            a.u[1] = *(const uint4*)(rowp + 16);
            FragBF bArr[4];
#pragma unroll
            for (int jj = 0; jj < 4; ++jj) {
                const int j = half * 4 + jj;
                const unsigned short* bp =
                    w1t + (size_t)(j * 16 + m) * 256 + kb + hi * 8;
                bArr[jj].u[0] = *(const uint4*)(bp);
                bArr[jj].u[1] = *(const uint4*)(bp + 16);
            }
#pragma unroll
            for (int jj = 0; jj < 4; ++jj) {
                acc[jj] = __builtin_amdgcn_wmma_f32_16x16x32_bf16(
                    false, a.v, false, bArr[jj].v, (short)0, acc[jj], false, false);
            }
        }
#pragma unroll
        for (int jj = 0; jj < 4; ++jj) {
            const int n = (half * 4 + jj) * 16 + m;
            const float bias = b1[n];
#pragma unroll
            for (int r = 0; r < 8; ++r) {
                const int mm = r + hi * 8;
                float v = acc[jj][r] + bias;
                sH[wave][mm * 128 + n] = f2bf(v > 0.f ? v : 0.f);
            }
        }
    }
    __syncthreads();

#pragma unroll
    for (int half = 0; half < 2; ++half) {
        v8f acc2[4];
#pragma unroll
        for (int jj = 0; jj < 4; ++jj) acc2[jj] = zf;
#pragma unroll
        for (int ks = 0; ks < 4; ++ks) {
            const int kb = ks * 32;
            const unsigned short* hp = &sH[wave][m * 128 + kb + aOff];
            FragBF a;
            a.u[0] = *(const uint4*)(hp);
            a.u[1] = *(const uint4*)(hp + 16);
            FragBF bArr[4];
#pragma unroll
            for (int jj = 0; jj < 4; ++jj) {
                const int j = half * 4 + jj;
                const unsigned short* bp =
                    w2t + (size_t)(j * 16 + m) * 128 + kb + hi * 8;
                bArr[jj].u[0] = *(const uint4*)(bp);
                bArr[jj].u[1] = *(const uint4*)(bp + 16);
            }
#pragma unroll
            for (int jj = 0; jj < 4; ++jj) {
                acc2[jj] = __builtin_amdgcn_wmma_f32_16x16x32_bf16(
                    false, a.v, false, bArr[jj].v, (short)0, acc2[jj], false, false);
            }
        }
        // x = nf + new + b2 -> LDS (f32)
#pragma unroll
        for (int r = 0; r < 8; ++r) {
            const int mm = r + hi * 8;
            int n2 = nBase + mm;
            if (n2 >= NNODES) n2 = NNODES - 1;
#pragma unroll
            for (int jj = 0; jj < 4; ++jj) {
                const int n = (half * 4 + jj) * 16 + m;
                sX[wave][mm * 128 + n] = nf[(size_t)n2 * 128 + n] + acc2[jj][r] + b2[n];
            }
        }
    }
    __syncthreads();

    // LayerNorm: lanes 0..15 each normalize one row
    if (lane < 16) {
        const int nd = nBase + lane;
        if (nd < NNODES) {
            float sum = 0.f, sq = 0.f;
            for (int k = 0; k < 128; ++k) {
                float v = sX[wave][lane * 128 + k];
                sum += v; sq += v * v;
            }
            const float mu  = sum * (1.f / 128.f);
            const float var = sq * (1.f / 128.f) - mu * mu;
            const float rs  = rsqrtf(var + 1e-5f);
            for (int k = 0; k < 128; ++k) {
                float y = (sX[wave][lane * 128 + k] - mu) * rs * lng[k] + lnb[k];
                nf[(size_t)nd * 128 + k] = y;
                xu[(size_t)nd * 256 + k] = f2bf(y);
            }
        }
    }
}

// ---------------------------------------------------------------------------
// Output projection: out = ReLU(nf @ W1 + b1) @ W2 + b2   (8 waves / block)
// ---------------------------------------------------------------------------
__global__ __launch_bounds__(256)
void k_output(const unsigned short* __restrict__ xu,
              const unsigned short* __restrict__ w1t, const float* __restrict__ b1,
              const unsigned short* __restrict__ w2t, const float* __restrict__ b2,
              float* __restrict__ out) {
    __shared__ unsigned short sH[8][16 * 128];
    const int tid  = threadIdx.x;
    const int wave = tid >> 5;
    const int lane = tid & 31;
    const int m    = lane & 15;
    const int hi   = lane >> 4;
    const int aOff = hi * 8;
    const int nBase = blockIdx.x * 128 + wave * 16;
    int node = nBase + m;
    if (node >= NNODES) node = NNODES - 1;

    const v8f zf = {};

#pragma unroll
    for (int half = 0; half < 2; ++half) {
        v8f acc[4];
#pragma unroll
        for (int jj = 0; jj < 4; ++jj) acc[jj] = zf;
#pragma unroll
        for (int ks = 0; ks < 4; ++ks) {
            const int kb = ks * 32;
            const unsigned short* rowp = xu + (size_t)node * 256 + kb + aOff;  // nf half
            FragBF a;
            a.u[0] = *(const uint4*)(rowp);
            a.u[1] = *(const uint4*)(rowp + 16);
            FragBF bArr[4];
#pragma unroll
            for (int jj = 0; jj < 4; ++jj) {
                const int j = half * 4 + jj;
                const unsigned short* bp =
                    w1t + (size_t)(j * 16 + m) * 128 + kb + hi * 8;
                bArr[jj].u[0] = *(const uint4*)(bp);
                bArr[jj].u[1] = *(const uint4*)(bp + 16);
            }
#pragma unroll
            for (int jj = 0; jj < 4; ++jj) {
                acc[jj] = __builtin_amdgcn_wmma_f32_16x16x32_bf16(
                    false, a.v, false, bArr[jj].v, (short)0, acc[jj], false, false);
            }
        }
#pragma unroll
        for (int jj = 0; jj < 4; ++jj) {
            const int n = (half * 4 + jj) * 16 + m;
            const float bias = b1[n];
#pragma unroll
            for (int r = 0; r < 8; ++r) {
                const int mm = r + hi * 8;
                float v = acc[jj][r] + bias;
                sH[wave][mm * 128 + n] = f2bf(v > 0.f ? v : 0.f);
            }
        }
    }
    __syncthreads();

#pragma unroll
    for (int half = 0; half < 2; ++half) {
        v8f acc2[4];
#pragma unroll
        for (int jj = 0; jj < 4; ++jj) acc2[jj] = zf;
#pragma unroll
        for (int ks = 0; ks < 4; ++ks) {
            const int kb = ks * 32;
            const unsigned short* hp = &sH[wave][m * 128 + kb + aOff];
            FragBF a;
            a.u[0] = *(const uint4*)(hp);
            a.u[1] = *(const uint4*)(hp + 16);
            FragBF bArr[4];
#pragma unroll
            for (int jj = 0; jj < 4; ++jj) {
                const int j = half * 4 + jj;
                const unsigned short* bp =
                    w2t + (size_t)(j * 16 + m) * 128 + kb + hi * 8;
                bArr[jj].u[0] = *(const uint4*)(bp);
                bArr[jj].u[1] = *(const uint4*)(bp + 16);
            }
#pragma unroll
            for (int jj = 0; jj < 4; ++jj) {
                acc2[jj] = __builtin_amdgcn_wmma_f32_16x16x32_bf16(
                    false, a.v, false, bArr[jj].v, (short)0, acc2[jj], false, false);
            }
        }
#pragma unroll
        for (int r = 0; r < 8; ++r) {
            const int mm = r + hi * 8;
            const int nd = nBase + mm;
            if (nd < NNODES) {
#pragma unroll
                for (int jj = 0; jj < 4; ++jj) {
                    const int n = (half * 4 + jj) * 16 + m;
                    out[(size_t)nd * 128 + n] = acc2[jj][r] + b2[n];
                }
            }
        }
    }
}

// ---------------------------------------------------------------------------
extern "C" void kernel_launch(void* const* d_in, const int* in_sizes, int n_in,
                              void* d_out, int out_size, void* d_ws, size_t ws_size,
                              hipStream_t stream) {
    const long long* eidx   = (const long long*)d_in[0];   // [2][E] int64
    const float* edge_attr  = (const float*)d_in[1];       // [E][3]
    const float* geo        = (const float*)d_in[2];       // [N][5]
    const float* ee_w1 = (const float*)d_in[4];
    const float* ee_b1 = (const float*)d_in[5];
    const float* ee_w2 = (const float*)d_in[6];
    const float* ee_b2 = (const float*)d_in[7];
    const float* ge_w  = (const float*)d_in[8];
    const float* ge_b  = (const float*)d_in[9];
    const float* msg_w1 = (const float*)d_in[10];   // [L][320][128]
    const float* msg_b1 = (const float*)d_in[11];
    const float* msg_w2 = (const float*)d_in[12];   // [L][128][128]
    const float* msg_b2 = (const float*)d_in[13];
    const float* upd_w1 = (const float*)d_in[14];   // [L][256][128]
    const float* upd_b1 = (const float*)d_in[15];
    const float* upd_w2 = (const float*)d_in[16];   // [L][128][128]
    const float* upd_b2 = (const float*)d_in[17];
    const float* ln_g   = (const float*)d_in[18];
    const float* ln_b   = (const float*)d_in[19];
    const float* op_w1  = (const float*)d_in[20];
    const float* op_b1  = (const float*)d_in[21];
    const float* op_w2  = (const float*)d_in[22];
    const float* op_b2  = (const float*)d_in[23];

    // workspace carve-out (256 B aligned)
    char* w = (char*)d_ws;
    size_t off = 0;
    auto take = [&](size_t bytes) -> void* {
        void* p = w + off;
        off = (off + bytes + 255) & ~(size_t)255;
        return p;
    };
    unsigned short* ef_bf = (unsigned short*)take((size_t)NEDGES * 64 * 2);
    unsigned short* xu    = (unsigned short*)take((size_t)NNODES * 256 * 2);
    float* nf  = (float*)take((size_t)NNODES * 128 * 4);
    float* agg = (float*)take((size_t)NNODES * 128 * 4);
    unsigned short* wm1t = (unsigned short*)take((size_t)NL * 128 * 320 * 2);
    unsigned short* wm2t = (unsigned short*)take((size_t)NL * 128 * 128 * 2);
    unsigned short* wu1t = (unsigned short*)take((size_t)NL * 128 * 256 * 2);
    unsigned short* wu2t = (unsigned short*)take((size_t)NL * 128 * 128 * 2);
    unsigned short* wo1t = (unsigned short*)take((size_t)128 * 128 * 2);
    unsigned short* wo2t = (unsigned short*)take((size_t)128 * 128 * 2);

    auto tr = [&](const float* src, unsigned short* dst, int K, int Nc) {
        int total = K * Nc;
        k_transpose_bf<<<(total + 255) / 256, 256, 0, stream>>>(src, dst, K, Nc);
    };
    for (int l = 0; l < NL; ++l) {
        tr(msg_w1 + (size_t)l * 320 * 128, wm1t + (size_t)l * 128 * 320, 320, 128);
        tr(msg_w2 + (size_t)l * 128 * 128, wm2t + (size_t)l * 128 * 128, 128, 128);
        tr(upd_w1 + (size_t)l * 256 * 128, wu1t + (size_t)l * 128 * 256, 256, 128);
        tr(upd_w2 + (size_t)l * 128 * 128, wu2t + (size_t)l * 128 * 128, 128, 128);
    }
    tr(op_w1, wo1t, 128, 128);
    tr(op_w2, wo2t, 128, 128);

    // encoders
    k_edge_enc<<<(NEDGES + 255) / 256, 256, 0, stream>>>(edge_attr, ee_w1, ee_b1,
                                                         ee_w2, ee_b2, ef_bf);
    k_geom<<<(NNODES + 255) / 256, 256, 0, stream>>>(geo, ge_w, ge_b, nf, xu);

    // GNN layers
    const int nAgg = NNODES * 128;
    for (int l = 0; l < NL; ++l) {
        k_zero<<<(nAgg + 255) / 256, 256, 0, stream>>>(agg, nAgg);
        k_message<<<NEDGES / (MSG_WAVES * 32), 128, 0, stream>>>(
            eidx, xu, ef_bf,
            wm1t + (size_t)l * 128 * 320, msg_b1 + (size_t)l * 128,
            wm2t + (size_t)l * 128 * 128, msg_b2 + (size_t)l * 128, agg);
        k_cvt_agg<<<(nAgg + 255) / 256, 256, 0, stream>>>(agg, xu);
        k_update<<<(NNODES + 63) / 64, 128, 0, stream>>>(
            xu, nf,
            wu1t + (size_t)l * 128 * 256, upd_b1 + (size_t)l * 128,
            wu2t + (size_t)l * 128 * 128, upd_b2 + (size_t)l * 128,
            ln_g + (size_t)l * 128, ln_b + (size_t)l * 128);
    }

    // output projection
    k_output<<<(NNODES + 127) / 128, 256, 0, stream>>>(xu, wo1t, op_b1, wo2t, op_b2,
                                                       (float*)d_out);
}
